// SlowMha_52218212385548
// MI455X (gfx1250) — matrix operand order repaired and checked by reading, not compiled, not run on hardware
//
#include <hip/hip_runtime.h>

// ---------------------------------------------------------------------------
// MHA forward for MI455X (gfx1250, wave32, WMMA).
// fp32 tensors in HBM, converted to bf16 while staging into LDS,
// v_wmma_f32_16x16x32_bf16 with fp32 accumulation, double-buffered LDS.
// ---------------------------------------------------------------------------

typedef __bf16 bf16_t;
typedef __attribute__((ext_vector_type(16))) __bf16 v16bf;
typedef __attribute__((ext_vector_type(8)))  float  v8f;

#define TILE_M 128
#define TILE_N 128
#define TILE_K 32
#define AS_STRIDE 40    // 32 + 8 pad bf16 -> 80B rows (16B aligned)
#define BS_STRIDE 136   // 128 + 8 pad bf16 -> 272B rows (16B aligned)

#define NEG_INF (-__builtin_inff())

// Pack 4 floats -> 4 bf16 and store as one 8-byte LDS store (ds_store_b64).
__device__ __forceinline__ void store4_bf16(bf16_t* dst, float4 v) {
  union { bf16_t h[4]; uint2 u; } p;
  p.h[0] = (bf16_t)v.x; p.h[1] = (bf16_t)v.y;
  p.h[2] = (bf16_t)v.z; p.h[3] = (bf16_t)v.w;
  *reinterpret_cast<uint2*>(dst) = p.u;
}

// D = A(M x K, lda) @ W(N x K, ldw)^T    (W transposed while staged into LDS)
// emode 0: plain store. emode 1: logits epilogue (scale + causal mask; blocks
//          above the diagonal are -inf filled without doing the GEMM; only
//          the diagonal block pays for the elementwise mask).
// blockIdx.z decodes (b,h) when H>0; per-(b,h) pointer offsets given as z*.
__global__ __launch_bounds__(256) void gemm_xt_kernel(
    const float* __restrict__ A, int lda,
    const float* __restrict__ W, int ldw,
    float* __restrict__ C, int ldc,
    int K, int emode, float scale, int H,
    long long zAb, long long zAh, long long zWb, long long zWh,
    long long zCb, long long zCh)
{
  const int bm = blockIdx.x, bn = blockIdx.y, z = blockIdx.z;
  const int b = (H > 0) ? (z / H) : 0;
  const int h = (H > 0) ? (z % H) : 0;
  A += zAb * b + zAh * h;
  W += zWb * b + zWh * h;
  C += zCb * b + zCh * h;

  const int tid  = threadIdx.x;
  const int wid  = tid >> 5;
  const int lane = tid & 31;
  const int wave_m = wid & 1;    // 2 waves over M (64 rows each)
  const int wave_n = wid >> 1;   // 4 waves over N (32 cols each)
  const int m0 = bm * TILE_M;
  const int n0 = bn * TILE_N;

  // Causal fast path: whole block above the diagonal -> -inf fill, no GEMM.
  if (emode == 1 && bn > bm) {
    float4 f4 = make_float4(NEG_INF, NEG_INF, NEG_INF, NEG_INF);
#pragma unroll
    for (int i = 0; i < 16; ++i) {
      int idx = i * 256 + tid;             // 4096 float4 in a 128x128 tile
      int r = idx >> 5;
      int c = (idx & 31) << 2;
      *reinterpret_cast<float4*>(&C[(size_t)(m0 + r) * ldc + n0 + c]) = f4;
    }
    return;
  }

  __shared__ bf16_t As[2][TILE_M * AS_STRIDE];
  __shared__ bf16_t Bs[2][TILE_K * BS_STRIDE];

  v8f acc[4][2];
  {
    v8f zero = {};
#pragma unroll
    for (int i = 0; i < 4; ++i)
#pragma unroll
      for (int j = 0; j < 2; ++j) acc[i][j] = zero;
  }

  const int lr = tid >> 3;         // 0..31  row group for staging
  const int lc = (tid & 7) * 4;    // 0..28  col (floats)

  auto stage = [&](int buf, int k0) {
    // A: 128x32 f32 -> bf16 LDS (row-major, packed 8B stores).
#pragma unroll
    for (int i = 0; i < 4; ++i) {
      int r = lr + i * 32;
      const float4 v = *reinterpret_cast<const float4*>(
          &A[(size_t)(m0 + r) * lda + k0 + lc]);
      store4_bf16(&As[buf][r * AS_STRIDE + lc], v);
    }
    // W transposed: Bs[k][n] = W[n0+n][k0+k] (inherent scatter).
#pragma unroll
    for (int i = 0; i < 4; ++i) {
      int n = lr + i * 32;
      const float4 v = *reinterpret_cast<const float4*>(
          &W[(size_t)(n0 + n) * ldw + k0 + lc]);
      Bs[buf][(lc + 0) * BS_STRIDE + n] = (bf16_t)v.x;
      Bs[buf][(lc + 1) * BS_STRIDE + n] = (bf16_t)v.y;
      Bs[buf][(lc + 2) * BS_STRIDE + n] = (bf16_t)v.z;
      Bs[buf][(lc + 3) * BS_STRIDE + n] = (bf16_t)v.w;
    }
  };

  stage(0, 0);
  int buf = 0;

  for (int k0 = 0; k0 < K; k0 += TILE_K) {
    __syncthreads();
    if (k0 + TILE_K < K) {
      if (k0 + 2 * TILE_K < K) {   // L2 prefetch of the tile after next
        __builtin_prefetch(&A[(size_t)(m0 + lr) * lda + k0 + 2 * TILE_K + lc], 0, 1);
        __builtin_prefetch(&W[(size_t)(n0 + lr) * ldw + k0 + 2 * TILE_K + lc], 0, 1);
      }
      stage(buf ^ 1, k0 + TILE_K); // overlaps with WMMAs below
    }

    // B fragments: lane L holds column N=L&15; lane<16 -> K 0..7 & 16..23; lane>=16 -> K 8..15 & 24..31.
    v16bf bfrag[2];
#pragma unroll
    for (int j = 0; j < 2; ++j) {
      const bf16_t* src = &Bs[buf][((lane >> 4) * 8) * BS_STRIDE + wave_n * 32 + j * 16 + (lane & 15)];
#pragma unroll
      for (int s = 0; s < 16; ++s)
        bfrag[j][s] = src[((s & 7) + (s >> 3) * 16) * BS_STRIDE];
    }
    // A fragments: lane<16 -> K 0..7 & 16..23; lane>=16 -> K 8..15 & 24..31.
    // All four loaded into distinct registers before the MAC burst so the
    // ds_loads pipeline ahead of the 8 back-to-back WMMAs.
    const int am = wave_m * 64 + (lane & 15);
    const int kh = (lane >> 4) * 8;
    v16bf afrag[4];
#pragma unroll
    for (int i = 0; i < 4; ++i) {
      const bf16_t* src = &As[buf][(am + i * 16) * AS_STRIDE + kh];
      uint4* dst = reinterpret_cast<uint4*>(&afrag[i]);
      dst[0] = *reinterpret_cast<const uint4*>(src);
      dst[1] = *reinterpret_cast<const uint4*>(src + 16);
    }
#pragma unroll
    for (int i = 0; i < 4; ++i)
#pragma unroll
      for (int j = 0; j < 2; ++j)
        acc[i][j] = __builtin_amdgcn_wmma_f32_16x16x32_bf16(
            false, afrag[i], false, bfrag[j], (short)0, acc[i][j], false, false);
    buf ^= 1;
  }

  // Epilogue: VGPR r <-> M = r (+8 for lanes 16..31); N = lane & 15.
  const int row_sel = (lane >> 4) * 8;
  const int ncol    = lane & 15;
  if (emode == 1 && bn == bm) {
    // Diagonal block: elementwise causal mask + scale.
#pragma unroll
    for (int i = 0; i < 4; ++i)
#pragma unroll
      for (int j = 0; j < 2; ++j)
#pragma unroll
        for (int r = 0; r < 8; ++r) {
          int mm = m0 + wave_m * 64 + i * 16 + r + row_sel;
          int nn = n0 + wave_n * 32 + j * 16 + ncol;
          float v = acc[i][j][r];
          C[(size_t)mm * ldc + nn] = (nn > mm) ? NEG_INF : v * scale;
        }
  } else {
    const float s = (emode == 1) ? scale : 1.0f;
#pragma unroll
    for (int i = 0; i < 4; ++i)
#pragma unroll
      for (int j = 0; j < 2; ++j)
#pragma unroll
        for (int r = 0; r < 8; ++r) {
          int mm = m0 + wave_m * 64 + i * 16 + r + row_sel;
          int nn = n0 + wave_n * 32 + j * 16 + ncol;
          C[(size_t)mm * ldc + nn] = acc[i][j][r] * s;
        }
  }
}

// O = P(attn, M x K f32) @ V(K x N f32), N = 128 (head dim), causal K cut.
__global__ __launch_bounds__(256) void gemm_pv_kernel(
    const float* __restrict__ A, int lda,
    const float* __restrict__ Bm, int ldb,
    float* __restrict__ C, int ldc,
    int K, int H,
    long long zAb, long long zAh, long long zBb, long long zBh,
    long long zCb, long long zCh)
{
  const int bm = blockIdx.x, z = blockIdx.z;
  const int b = z / H, h = z % H;
  A  += zAb * b + zAh * h;
  Bm += zBb * b + zBh * h;
  C  += zCb * b + zCh * h;

  const int tid  = threadIdx.x;
  const int wid  = tid >> 5;
  const int lane = tid & 31;
  const int wave_m = wid & 1;
  const int wave_n = wid >> 1;
  const int m0 = bm * TILE_M;

  __shared__ bf16_t As[2][TILE_M * AS_STRIDE];
  __shared__ bf16_t Bs[2][TILE_K * BS_STRIDE];

  v8f acc[4][2];
  {
    v8f zero = {};
#pragma unroll
    for (int i = 0; i < 4; ++i)
#pragma unroll
      for (int j = 0; j < 2; ++j) acc[i][j] = zero;
  }

  const int lr = tid >> 3;
  const int lc = (tid & 7) * 4;
  const int kk = tid >> 3;          // B stage: row k
  const int cc = (tid & 7) * 16;    // B stage: 16 cols per thread

  const int kcausal = (bm + 1) * TILE_M;       // keys beyond this are all zero
  const int kend = K < kcausal ? K : kcausal;  // multiple of 32

  auto stage = [&](int buf, int k0) {
    // A (attention probs) f32 -> bf16, packed stores.
#pragma unroll
    for (int i = 0; i < 4; ++i) {
      int r = lr + i * 32;
      const float4 v = *reinterpret_cast<const float4*>(
          &A[(size_t)(m0 + r) * lda + k0 + lc]);
      store4_bf16(&As[buf][r * AS_STRIDE + lc], v);
    }
    // V directly (already K-major): Bs[k][n] = V[k0+k][n], packed stores.
#pragma unroll
    for (int q = 0; q < 4; ++q) {
      const float4 v = *reinterpret_cast<const float4*>(
          &Bm[(size_t)(k0 + kk) * ldb + cc + q * 4]);
      store4_bf16(&Bs[buf][kk * BS_STRIDE + cc + q * 4], v);
    }
  };

  stage(0, 0);
  int buf = 0;

  for (int k0 = 0; k0 < kend; k0 += TILE_K) {
    __syncthreads();
    if (k0 + TILE_K < kend) {
      if (k0 + 2 * TILE_K < kend) {
        __builtin_prefetch(&A[(size_t)(m0 + lr) * lda + k0 + 2 * TILE_K + lc], 0, 1);
        __builtin_prefetch(&Bm[(size_t)(k0 + 2 * TILE_K + kk) * ldb + cc], 0, 1);
      }
      stage(buf ^ 1, k0 + TILE_K);
    }

    v16bf bfrag[2];
#pragma unroll
    for (int j = 0; j < 2; ++j) {
      const bf16_t* src = &Bs[buf][((lane >> 4) * 8) * BS_STRIDE + wave_n * 32 + j * 16 + (lane & 15)];
#pragma unroll
      for (int s = 0; s < 16; ++s)
        bfrag[j][s] = src[((s & 7) + (s >> 3) * 16) * BS_STRIDE];
    }
    const int am = wave_m * 64 + (lane & 15);
    const int kh = (lane >> 4) * 8;
    v16bf afrag[4];
#pragma unroll
    for (int i = 0; i < 4; ++i) {
      const bf16_t* src = &As[buf][(am + i * 16) * AS_STRIDE + kh];
      uint4* dst = reinterpret_cast<uint4*>(&afrag[i]);
      dst[0] = *reinterpret_cast<const uint4*>(src);
      dst[1] = *reinterpret_cast<const uint4*>(src + 16);
    }
#pragma unroll
    for (int i = 0; i < 4; ++i)
#pragma unroll
      for (int j = 0; j < 2; ++j)
        acc[i][j] = __builtin_amdgcn_wmma_f32_16x16x32_bf16(
            false, afrag[i], false, bfrag[j], (short)0, acc[i][j], false, false);
    buf ^= 1;
  }

  const int row_sel = (lane >> 4) * 8;
  const int ncol    = lane & 15;
#pragma unroll
  for (int i = 0; i < 4; ++i)
#pragma unroll
    for (int j = 0; j < 2; ++j)
#pragma unroll
      for (int r = 0; r < 8; ++r) {
        int mm = m0 + wave_m * 64 + i * 16 + r + row_sel;
        int nn = wave_n * 32 + j * 16 + ncol;
        C[(size_t)mm * ldc + nn] = acc[i][j][r];
      }
}

// In-place RoPE on Q and K parts of qkv workspace (B,T,3,H,128).
__global__ void rope_kernel(float* __restrict__ qkv,
                            const float* __restrict__ cosT,
                            const float* __restrict__ sinT)
{
  int idx = blockIdx.x * blockDim.x + threadIdx.x;   // 2^22 rotation pairs
  int i = idx & 63;
  int h = (idx >> 6) & 7;
  int a = (idx >> 9) & 1;
  int t = (idx >> 10) & 2047;
  int b = idx >> 21;
  size_t base = ((size_t)(b * 2048 + t) * 3 + a) * 1024 + h * 128 + 2 * i;
  float re = qkv[base], im = qkv[base + 1];
  float c = cosT[t * 64 + i], s = sinT[t * 64 + i];
  qkv[base]     = re * c - im * s;
  qkv[base + 1] = re * s + im * c;
}

// Row softmax over 2048-wide rows of attn (in-place). -inf -> 0.
// wave32 shuffle reduction: 2 barriers per row instead of 16.
__global__ __launch_bounds__(256) void softmax_kernel(float* __restrict__ attn)
{
  float* p = attn + (size_t)blockIdx.x * 2048;
  __shared__ float redm[8];
  __shared__ float reds[8];
  const int tid  = threadIdx.x;
  const int wid  = tid >> 5;
  const int lane = tid & 31;

  float vals[8];
  float m = NEG_INF;
#pragma unroll
  for (int i = 0; i < 8; ++i) { vals[i] = p[tid + i * 256]; m = fmaxf(m, vals[i]); }
#pragma unroll
  for (int off = 16; off > 0; off >>= 1) m = fmaxf(m, __shfl_xor(m, off, 32));
  if (lane == 0) redm[wid] = m;
  __syncthreads();
  m = redm[0];
#pragma unroll
  for (int i = 1; i < 8; ++i) m = fmaxf(m, redm[i]);

  float sum = 0.f;
#pragma unroll
  for (int i = 0; i < 8; ++i) { vals[i] = __expf(vals[i] - m); sum += vals[i]; }
#pragma unroll
  for (int off = 16; off > 0; off >>= 1) sum += __shfl_xor(sum, off, 32);
  if (lane == 0) reds[wid] = sum;
  __syncthreads();
  sum = reds[0];
#pragma unroll
  for (int i = 1; i < 8; ++i) sum += reds[i];

  float inv = 1.0f / sum;
#pragma unroll
  for (int i = 0; i < 8; ++i) p[tid + i * 256] = vals[i] * inv;
}

extern "C" void kernel_launch(void* const* d_in, const int* in_sizes, int n_in,
                              void* d_out, int out_size, void* d_ws, size_t ws_size,
                              hipStream_t stream)
{
  (void)in_sizes; (void)n_in; (void)out_size; (void)ws_size;
  const float* x      = (const float*)d_in[0];   // (2,2048,1024)
  const float* qkv_w  = (const float*)d_in[1];   // (3,1024,1024)
  const float* w_out  = (const float*)d_in[2];   // (1024,1024)
  const float* cosT   = (const float*)d_in[3];   // (2048,64)
  const float* sinT   = (const float*)d_in[4];   // (2048,64)

  float* out  = (float*)d_out;                           // (2,2048,1024)
  float* attn = (float*)d_out + (size_t)2 * 2048 * 1024; // (2,8,2048,2048)

  float* qkv  = (float*)d_ws;                            // (2,2048,3,1024) 48MB
  float* Obuf = qkv + (size_t)2 * 2048 * 3072;           // (2,2048,1024)   16MB

  const long long T = 2048;
  const float scale = 0.08838834764831845f;  // 128^-0.5

  // 1) QKV projection: (4096 x 1024) @ (3072 x 1024)^T
  gemm_xt_kernel<<<dim3(32, 24, 1), 256, 0, stream>>>(
      x, 1024, qkv_w, 1024, qkv, 3072, 1024, 0, 1.0f, 0,
      0, 0, 0, 0, 0, 0);

  // 2) RoPE on Q and K
  rope_kernel<<<16384, 256, 0, stream>>>(qkv, cosT, sinT);

  // 3) logits = scale * Q @ K^T with causal mask, per (b,h)
  gemm_xt_kernel<<<dim3(16, 16, 16), 256, 0, stream>>>(
      qkv, 3072, qkv + 1024, 3072, attn, 2048, 128, 1, scale, 8,
      T * 3072, 128, T * 3072, 128,
      8 * T * T, T * T);

  // 4) row softmax over attn (in-place), 2*8*2048 rows
  softmax_kernel<<<32768, 256, 0, stream>>>(attn);

  // 5) O = attn @ V, per (b,h), causal K cut
  gemm_pv_kernel<<<dim3(16, 1, 16), 256, 0, stream>>>(
      attn, 2048, qkv + 2048, 3072, Obuf, 1024, 2048, 8,
      8 * T * T, T * T, T * 3072, 128,
      T * 1024, 128);

  // 6) out = O @ w_out^T
  gemm_xt_kernel<<<dim3(32, 8, 1), 256, 0, stream>>>(
      Obuf, 1024, w_out, 1024, out, 1024, 1024, 0, 1.0f, 0,
      0, 0, 0, 0, 0, 0);
}
